// GroupedQueryAttention_28776280883984
// MI455X (gfx1250) — matrix operand order repaired and checked by reading, not compiled, not run on hardware
//
#include <hip/hip_runtime.h>
#include <hip/hip_bf16.h>
#include <math.h>

typedef __attribute__((ext_vector_type(16))) _Float16 v16h;
typedef __attribute__((ext_vector_type(8)))  float    v8f;
typedef unsigned int u32x4 __attribute__((ext_vector_type(4)));
typedef unsigned int u32x8 __attribute__((ext_vector_type(8)));

namespace {
constexpr int NB  = 2;     // batch
constexpr int NT  = 2048;  // sequence length
constexpr int NC  = 2048;  // model dim
constexpr int NH  = 16;    // query heads
constexpr int NKV = 4;     // kv heads
constexpr int DH  = 128;   // head dim
constexpr int NM  = NB * NT;            // 4096 GEMM rows
constexpr float SM_SCALE = 0.25f;       // 1/sqrt(H) per reference
}

union F16x16 { v16h v; unsigned int u[8]; };

// ------------------------------------------------------------------ TDM
// Issue a 2D tensor_load_to_lds: tile (tile1 rows x tile0 elems) of a
// row-major f16 tensor (row stride = stride0 elems) -> LDS at lds_off with
// pad_amount(+1) dwords of padding appended after every 2^(padi+1) dwords.
// D# layout per CDNA5 ISA ch.8: group0 = {count|flags, lds_addr,
// global_addr[31:0], global_addr[56:32]|type<<30}; group1 packs data_size,
// pad ctrl, tensor dims/strides and tile dims.
__device__ inline void tdm_load_2d(const _Float16* gptr, unsigned lds_off,
                                   unsigned dim0, unsigned dim1,
                                   unsigned stride0,
                                   unsigned tile0, unsigned tile1,
                                   unsigned padi, unsigned pada) {
  unsigned long long ga = (unsigned long long)(size_t)gptr;
  u32x4 g0;
  g0[0] = 1u;                                   // count=1, user descriptor
  g0[1] = lds_off;                              // LDS byte address
  g0[2] = (unsigned)(ga & 0xFFFFFFFFu);         // global_addr[31:0]
  g0[3] = (unsigned)((ga >> 32) & 0x1FFFFFFu)   // global_addr[56:32]
          | (2u << 30);                         // type=2 ("image")
  u32x8 g1;
  g1[0] = (1u << 16)                            // data_size = 2 bytes
        | (1u << 20)                            // pad_enable
        | (padi << 22)                          // pad_interval
        | (pada << 25);                         // pad_amount
  g1[1] = (dim0 & 0xFFFFu) << 16;               // tensor_dim0[15:0]
  g1[2] = (dim0 >> 16) | ((dim1 & 0xFFFFu) << 16); // dim0[31:16], dim1[15:0]
  g1[3] = (dim1 >> 16) | (tile0 << 16);         // dim1[31:16], tile_dim0
  g1[4] = tile1;                                // tile_dim1 (tile_dim2 = 0)
  g1[5] = stride0;                              // tensor_dim0_stride[31:0]
  g1[6] = 0u;                                   // stride0[47:32], stride1 lo
  g1[7] = 0u;                                   // stride1 hi
  asm volatile("tensor_load_to_lds %0, %1" :: "s"(g0), "s"(g1) : "memory");
}

// ---------------------------------------------------------------- converts
__global__ void cvt_f32_f16(const float* __restrict__ src,
                            _Float16* __restrict__ dst, int n) {
  int i = blockIdx.x * blockDim.x + threadIdx.x;
  if (i < n) dst[i] = (_Float16)src[i];
}

// W [K][N] f32 -> Wt [N][K] f16 (write-coalesced)
__global__ void cvt_wT(const float* __restrict__ in, _Float16* __restrict__ out,
                       int K, int N) {
  int idx = blockIdx.x * blockDim.x + threadIdx.x;
  if (idx >= K * N) return;
  int k = idx % K, n = idx / K;
  out[(size_t)n * K + k] = (_Float16)in[(size_t)k * N + n];
}

// v f32 [B][T][NKV][DH] -> f16 [B][NKV][DH][T] (write-coalesced)
__global__ void cvt_vT(const float* __restrict__ in, _Float16* __restrict__ out) {
  int idx = blockIdx.x * blockDim.x + threadIdx.x;
  if (idx >= NB * NKV * DH * NT) return;
  int t = idx % NT;
  int r = idx / NT;
  int d = r % DH; r /= DH;
  int h = r % NKV;
  int b = r / NKV;
  float val = in[((size_t)(b * NT + t) * NKV + h) * DH + d];
  out[((size_t)(b * NKV + h) * DH + d) * NT + t] = (_Float16)val;
}

// ---------------------------------------------------------------- GEMM
// C[M,N] = A[M,K](f16 row-major) * Wt[N,K](f16 row-major, i.e. W^T) + bias.
// 256 threads / 8 waves, 128x128 tile, K-step 32, double-buffered TDM loads.
__global__ __launch_bounds__(256)
void gemm_f16_wmma(const _Float16* __restrict__ A, const _Float16* __restrict__ Wt,
                   const float* __restrict__ bias, float* __restrict__ C,
                   int M, int N, int K) {
  __shared__ _Float16 As[2][128][40];   // [m][k], 4-dword pad per 16-dword row
  __shared__ _Float16 Bs[2][128][40];   // [n][k]

  const int tid  = threadIdx.x;
  const int lane = tid & 31;
  const int w    = tid >> 5;
  const int wr   = w >> 1;           // 0..3  (32 rows each)
  const int wc   = w & 1;            // 0..1  (64 cols each)
  const int half = lane >> 4;
  const int l16  = lane & 15;
  const int m0 = blockIdx.y * 128, n0 = blockIdx.x * 128;
  const int nsteps = K / 32;
  const bool leader = (tid < 32);

  auto issue = [&](int s) {
    const int buf = s & 1;
    tdm_load_2d(A  + (size_t)m0 * K + s * 32,
                (unsigned)(size_t)(&As[buf][0][0]),
                (unsigned)K, (unsigned)M, (unsigned)K, 32u, 128u, 3u, 3u);
    tdm_load_2d(Wt + (size_t)n0 * K + s * 32,
                (unsigned)(size_t)(&Bs[buf][0][0]),
                (unsigned)K, (unsigned)N, (unsigned)K, 32u, 128u, 3u, 3u);
  };
  if (leader) { issue(0); if (nsteps > 1) issue(1); }

  v8f zero = {};
  v8f acc[2][4];
#pragma unroll
  for (int mt = 0; mt < 2; ++mt)
#pragma unroll
    for (int nt = 0; nt < 4; ++nt) acc[mt][nt] = zero;

  for (int s = 0; s < nsteps; ++s) {
    if (s + 1 < nsteps) __builtin_amdgcn_s_wait_tensorcnt(2);
    else                __builtin_amdgcn_s_wait_tensorcnt(0);
    __syncthreads();
    const int buf = s & 1;

    const int kbA = half * 8;
    const int kbB = half * 8;
    F16x16 afr[2], bfr[4];
#pragma unroll
    for (int mt = 0; mt < 2; ++mt) {
      int m = wr * 32 + mt * 16 + l16;
#pragma unroll
      for (int j = 0; j < 8; ++j) {
        int kk = (j < 4) ? (kbA + 2 * j) : (16 + kbA + 2 * (j - 4));
        afr[mt].u[j] = *reinterpret_cast<const unsigned int*>(&As[buf][m][kk]);
      }
    }
#pragma unroll
    for (int nt = 0; nt < 4; ++nt) {
      int n = wc * 64 + nt * 16 + l16;
#pragma unroll
      for (int j = 0; j < 8; ++j)
        bfr[nt].u[j] =
            *reinterpret_cast<const unsigned int*>(&Bs[buf][n][(j < 4) ? (kbB + 2 * j) : (16 + kbB + 2 * (j - 4))]);
    }
#pragma unroll
    for (int mt = 0; mt < 2; ++mt)
#pragma unroll
      for (int nt = 0; nt < 4; ++nt)
        acc[mt][nt] = __builtin_amdgcn_wmma_f32_16x16x32_f16(
            false, afr[mt].v, false, bfr[nt].v, (short)0, acc[mt][nt],
            false, false);
    __syncthreads();
    if (leader && s + 2 < nsteps) issue(s + 2);
  }

#pragma unroll
  for (int mt = 0; mt < 2; ++mt)
#pragma unroll
    for (int nt = 0; nt < 4; ++nt) {
      int col = n0 + wc * 64 + nt * 16 + l16;
      float bv = bias ? bias[col] : 0.0f;
#pragma unroll
      for (int j = 0; j < 8; ++j) {
        int row = m0 + wr * 32 + mt * 16 + half * 8 + j;
        C[(size_t)row * N + col] = acc[mt][nt][j] + bv;
      }
    }
}

// ---------------------------------------------------------------- RoPE
// in:  f32 [B][T][nh][DH]   out: f16 [B][nh][T][DH], optional rotation
__global__ void rope_cvt(const float* __restrict__ in, _Float16* __restrict__ out,
                         int nheads, int dorope) {
  const int hd2 = DH / 2;
  int idx = blockIdx.x * blockDim.x + threadIdx.x;
  int total = NB * NT * nheads * hd2;
  if (idx >= total) return;
  int i = idx % hd2;
  int r = idx / hd2;
  int h = r % nheads; r /= nheads;
  int t = r % NT;
  int b = r / NT;
  const float2 ab = *reinterpret_cast<const float2*>(
      in + ((size_t)(b * NT + t) * nheads + h) * DH + 2 * i);
  float oa = ab.x, ob = ab.y;
  if (dorope) {
    float invf = __powf(10000.0f, -(float)(2 * i) / (float)DH);
    float ang = (float)t * invf;
    float s, c;
    __sincosf(ang, &s, &c);
    oa = ab.x * c - ab.y * s;
    ob = ab.x * s + ab.y * c;
  }
  union { unsigned int u; _Float16 h2[2]; } pk;
  pk.h2[0] = (_Float16)oa; pk.h2[1] = (_Float16)ob;
  *reinterpret_cast<unsigned int*>(
      out + ((size_t)(b * nheads + h) * NT + t) * DH + 2 * i) = pk.u;
}

// ---------------------------------------------------------------- attention
// q: [B][NH][T][DH] f16, k: [B][NKV][T][DH] f16, vT: [B][NKV][DH][T] f16,
// o: [B][T][NH*DH] f16. Block = 128 threads (4 waves), 64 query rows,
// key blocks of 32, TDM double-buffered K / V^T staging.
__global__ __launch_bounds__(128)
void flash_attn(const _Float16* __restrict__ q, const _Float16* __restrict__ k,
                const _Float16* __restrict__ vt, _Float16* __restrict__ o) {
  __shared__ _Float16 Ks[2][32][132];    // [key][dim] (+2-dword pad per row)
  __shared__ _Float16 Vts[2][128][36];   // [dim][key] (+2-dword pad per row)
  __shared__ _Float16 Ps[4][16][32];     // per-wave P scratch

  const int tid  = threadIdx.x;
  const int lane = tid & 31, w = tid >> 5;
  const int half = lane >> 4, l16 = lane & 15;
  const int bh = blockIdx.y;
  const int b = bh / NH, h = bh % NH;
  const int kvh = h / (NH / NKV);
  const int qblock = blockIdx.x * 64;
  const int qrow0 = qblock + w * 16;
  const bool leader = (tid < 32);

  const _Float16* qbase  = q  + (size_t)(b * NH + h) * NT * DH;
  const _Float16* kbase  = k  + (size_t)(b * NKV + kvh) * NT * DH;
  const _Float16* vtbase = vt + (size_t)(b * NKV + kvh) * DH * NT;

  const int kend = qblock + 64;          // causal bound for this block
  const int nsteps = kend / 32;

  auto issue = [&](int s) {
    const int buf = s & 1;
    // K tile: 32 keys x 128 dims, row-major stride DH
    tdm_load_2d(kbase + (size_t)(s * 32) * DH,
                (unsigned)(size_t)(&Ks[buf][0][0]),
                (unsigned)DH, (unsigned)NT, (unsigned)DH, 128u, 32u, 5u, 1u);
    // V^T tile: 128 dims x 32 keys, row-major stride NT
    tdm_load_2d(vtbase + s * 32,
                (unsigned)(size_t)(&Vts[buf][0][0]),
                (unsigned)NT, (unsigned)DH, (unsigned)NT, 32u, 128u, 3u, 1u);
  };
  if (leader) { issue(0); if (nsteps > 1) issue(1); }

  // Q fragments: 16 rows x 128 dims = 4 A-frags of 16x32
  F16x16 qa[4];
  {
    const int m = qrow0 + l16;
    const int kb = half * 8;
#pragma unroll
    for (int c = 0; c < 4; ++c)
#pragma unroll
      for (int j = 0; j < 8; ++j) {
        int kk = c * 32 + ((j < 4) ? (kb + 2 * j) : (16 + kb + 2 * (j - 4)));
        qa[c].u[j] = *reinterpret_cast<const unsigned int*>(
            qbase + (size_t)m * DH + kk);
      }
  }

  v8f zero = {};
  v8f O[8];
#pragma unroll
  for (int n = 0; n < 8; ++n) O[n] = zero;
  float mrow[8], lrow[8];
#pragma unroll
  for (int j = 0; j < 8; ++j) { mrow[j] = -INFINITY; lrow[j] = 0.0f; }

  for (int s = 0; s < nsteps; ++s) {
    if (s + 1 < nsteps) __builtin_amdgcn_s_wait_tensorcnt(2);
    else                __builtin_amdgcn_s_wait_tensorcnt(0);
    __syncthreads();
    const int buf = s & 1;
    const int kb0 = s * 32;

    // S = Q @ K^T : two 16x16 score tiles (keys kb0..+15, kb0+16..+31)
    v8f S[2];
#pragma unroll
    for (int st = 0; st < 2; ++st) {
      S[st] = zero;
      const int key = st * 16 + l16;
      const int kb2 = half * 8;
#pragma unroll
      for (int c = 0; c < 4; ++c) {
        F16x16 bf;
#pragma unroll
        for (int j = 0; j < 8; ++j)
          bf.u[j] = *reinterpret_cast<const unsigned int*>(
              &Ks[buf][key][c * 32 + ((j < 4) ? (kb2 + 2 * j) : (16 + kb2 + 2 * (j - 4)))]);
        S[st] = __builtin_amdgcn_wmma_f32_16x16x32_f16(
            false, qa[c].v, false, bf.v, (short)0, S[st], false, false);
      }
    }

    // scale + causal mask + online softmax (C-layout: row = j + 8*half)
    float p0[8], p1[8], alpha[8];
#pragma unroll
    for (int j = 0; j < 8; ++j) {
      int r = qrow0 + half * 8 + j;
      float s0 = S[0][j] * SM_SCALE;
      float s1 = S[1][j] * SM_SCALE;
      if (kb0 + l16 > r)      s0 = -INFINITY;
      if (kb0 + 16 + l16 > r) s1 = -INFINITY;
      float mv = fmaxf(s0, s1);
#pragma unroll
      for (int d = 1; d < 16; d <<= 1) mv = fmaxf(mv, __shfl_xor(mv, d, 16));
      float mnew = fmaxf(mrow[j], mv);
      alpha[j] = __expf(mrow[j] - mnew);
      p0[j] = __expf(s0 - mnew);
      p1[j] = __expf(s1 - mnew);
      float rs = p0[j] + p1[j];
#pragma unroll
      for (int d = 1; d < 16; d <<= 1) rs += __shfl_xor(rs, d, 16);
      lrow[j] = lrow[j] * alpha[j] + rs;
      mrow[j] = mnew;
    }
#pragma unroll
    for (int n = 0; n < 8; ++n)
#pragma unroll
      for (int j = 0; j < 8; ++j) O[n][j] *= alpha[j];

    // P (C-layout f32) -> per-wave LDS -> A-fragment f16
#pragma unroll
    for (int j = 0; j < 8; ++j) {
      int rloc = half * 8 + j;
      Ps[w][rloc][l16]      = (_Float16)p0[j];
      Ps[w][rloc][16 + l16] = (_Float16)p1[j];
    }
    asm volatile("s_wait_dscnt 0x0" ::: "memory");
    F16x16 pf;
    {
      const int kb = half * 8;
#pragma unroll
      for (int j = 0; j < 8; ++j) {
        int kk = (j < 4) ? (kb + 2 * j) : (16 + kb + 2 * (j - 4));
        pf.u[j] = *reinterpret_cast<const unsigned int*>(&Ps[w][l16][kk]);
      }
    }
    // O += P @ V (8 dim-tiles of 16)
#pragma unroll
    for (int n = 0; n < 8; ++n) {
      const int dim = n * 16 + l16;
      const int kb2 = half * 8;
      F16x16 bf;
#pragma unroll
      for (int j = 0; j < 8; ++j)
        bf.u[j] = *reinterpret_cast<const unsigned int*>(
            &Vts[buf][dim][(j < 4) ? (kb2 + 2 * j) : (16 + kb2 + 2 * (j - 4))]);
      O[n] = __builtin_amdgcn_wmma_f32_16x16x32_f16(
          false, pf.v, false, bf.v, (short)0, O[n], false, false);
    }
    __syncthreads();
    if (leader && s + 2 < nsteps) issue(s + 2);
  }

  // normalize and store to [B][T][NH*DH] f16
#pragma unroll
  for (int n = 0; n < 8; ++n)
#pragma unroll
    for (int j = 0; j < 8; ++j) {
      int r = qrow0 + half * 8 + j;
      int dcol = n * 16 + l16;
      float val = O[n][j] / lrow[j];
      o[(size_t)(b * NT + r) * (NH * DH) + h * DH + dcol] = (_Float16)val;
    }
}

// ---------------------------------------------------------------- launch
extern "C" void kernel_launch(void* const* d_in, const int* in_sizes, int n_in,
                              void* d_out, int out_size, void* d_ws, size_t ws_size,
                              hipStream_t stream) {
  (void)in_sizes; (void)n_in; (void)out_size; (void)ws_size;
  const float* x  = (const float*)d_in[0];
  const float* Wq = (const float*)d_in[1];
  const float* bq = (const float*)d_in[2];
  const float* Wk = (const float*)d_in[3];
  const float* bk = (const float*)d_in[4];
  const float* Wv = (const float*)d_in[5];
  const float* bv = (const float*)d_in[6];
  const float* Wo = (const float*)d_in[7];
  const float* bo = (const float*)d_in[8];
  float* out = (float*)d_out;

  char* ws = (char*)d_ws;
  size_t off = 0;
  auto take = [&](size_t bytes) {
    char* p = ws + off;
    off += (bytes + 255) & ~(size_t)255;
    return p;
  };

  _Float16* x16   = (_Float16*)take((size_t)NM * NC * 2);
  _Float16* Wq16t = (_Float16*)take((size_t)NC * NH * DH * 2);
  _Float16* Wk16t = (_Float16*)take((size_t)NC * NKV * DH * 2);
  _Float16* Wv16t = (_Float16*)take((size_t)NC * NKV * DH * 2);
  _Float16* Wo16t = (_Float16*)take((size_t)NH * DH * NC * 2);
  float* qf = (float*)take((size_t)NM * NH * DH * 4);
  float* kf = (float*)take((size_t)NM * NKV * DH * 4);
  float* vf = (float*)take((size_t)NM * NKV * DH * 4);
  _Float16* q16  = (_Float16*)take((size_t)NB * NH * NT * DH * 2);
  _Float16* k16  = (_Float16*)take((size_t)NB * NKV * NT * DH * 2);
  _Float16* v16t = (_Float16*)take((size_t)NB * NKV * DH * NT * 2);
  _Float16* a16  = (_Float16*)qf;  // reuse q-f32 region once RoPE consumed it

  cvt_f32_f16<<<(unsigned)(((size_t)NM * NC + 255) / 256), 256, 0, stream>>>(
      x, x16, NM * NC);
  cvt_wT<<<(unsigned)(((size_t)NC * NH * DH + 255) / 256), 256, 0, stream>>>(
      Wq, Wq16t, NC, NH * DH);
  cvt_wT<<<(unsigned)(((size_t)NC * NKV * DH + 255) / 256), 256, 0, stream>>>(
      Wk, Wk16t, NC, NKV * DH);
  cvt_wT<<<(unsigned)(((size_t)NC * NKV * DH + 255) / 256), 256, 0, stream>>>(
      Wv, Wv16t, NC, NKV * DH);
  cvt_wT<<<(unsigned)(((size_t)NH * DH * NC + 255) / 256), 256, 0, stream>>>(
      Wo, Wo16t, NH * DH, NC);

  dim3 blk(256);
  gemm_f16_wmma<<<dim3((NH * DH) / 128,  NM / 128), blk, 0, stream>>>(
      x16, Wq16t, bq, qf, NM, NH * DH, NC);
  gemm_f16_wmma<<<dim3((NKV * DH) / 128, NM / 128), blk, 0, stream>>>(
      x16, Wk16t, bk, kf, NM, NKV * DH, NC);
  gemm_f16_wmma<<<dim3((NKV * DH) / 128, NM / 128), blk, 0, stream>>>(
      x16, Wv16t, bv, vf, NM, NKV * DH, NC);

  int nq  = NB * NT * NH  * (DH / 2);
  int nkv = NB * NT * NKV * (DH / 2);
  rope_cvt<<<(nq  + 255) / 256, 256, 0, stream>>>(qf, q16, NH, 1);
  rope_cvt<<<(nkv + 255) / 256, 256, 0, stream>>>(kf, k16, NKV, 1);
  cvt_vT<<<(NB * NKV * DH * NT + 255) / 256, 256, 0, stream>>>(vf, v16t);

  flash_attn<<<dim3(NT / 64, NB * NH), 128, 0, stream>>>(q16, k16, v16t, a16);

  gemm_f16_wmma<<<dim3(NC / 128, NM / 128), blk, 0, stream>>>(
      a16, Wo16t, bo, out, NM, NC, NH * DH);
}
